// VectorQuantizer_16896401343122
// MI455X (gfx1250) — matrix operand order repaired and checked by reading, not compiled, not run on hardware
//
#include <hip/hip_runtime.h>
#include <cstdint>

// Problem constants (from reference): B=32, C=64, H=64, W=64, K=512
#define HW_    4096        // H*W
#define CHW_   262144      // C*H*W
#define NUMEL_ 8388608     // B*C*H*W
#define NROWS_ 131072      // B*H*W

typedef __attribute__((ext_vector_type(16))) __bf16 bf16x16;
typedef __attribute__((ext_vector_type(8)))  __bf16 bf16x8;
typedef __attribute__((ext_vector_type(8)))  float  f32x8;

// Matches the builtin's parameter type: generic pointer to a 16-byte int vector.
typedef int v4i __attribute__((__vector_size__(16)));

#if defined(__has_builtin)
#if __has_builtin(__builtin_amdgcn_global_load_async_to_lds_b128)
#define HAVE_ASYNC_LDS 1
#endif
#endif

// Workspace layout (bytes):
//  [0,      65536)  cb_hi   : 32768 bf16
//  [65536, 131072)  cb_lo   : 32768 bf16   (contiguous with cb_hi for 1-shot staging)
//  [131072,133120)  cnorm   : 512 f32
//  [133120,137216)  partial : 1024 f32

__global__ __launch_bounds__(256) void vq_prep_split(const float* __restrict__ cb,
                                                     __bf16* __restrict__ hi,
                                                     __bf16* __restrict__ lo) {
    int i = blockIdx.x * 256 + threadIdx.x;      // 0..32767
    float x = cb[i];
    __bf16 h = (__bf16)x;
    hi[i] = h;
    lo[i] = (__bf16)(x - (float)h);
}

__global__ __launch_bounds__(256) void vq_prep_norm(const float* __restrict__ cb,
                                                    float* __restrict__ cnorm) {
    int k = blockIdx.x * 256 + threadIdx.x;      // 0..511
    float s = 0.f;
    #pragma unroll
    for (int c = 0; c < 64; ++c) { float v = cb[k * 64 + c]; s = fmaf(v, v, s); }
    cnorm[k] = s;
}

// Build a 16-element bf16 A/B fragment from two aligned 16B LDS chunks at p and p+16.
static __device__ inline bf16x16 ld_frag16(const __bf16* p) {
    bf16x8 a = *(const bf16x8*)p;
    bf16x8 b = *(const bf16x8*)(p + 16);
    return __builtin_shufflevector(a, b, 0,1,2,3,4,5,6,7,8,9,10,11,12,13,14,15);
}

// Main kernel: 256 threads = 8 waves; each wave owns 16 z-rows; block owns 128 rows.
// LDS: codebook hi/lo (bf16) + cnorm + per-block indices + reduction buffer.
__global__ __launch_bounds__(256) void vq_main(const float* __restrict__ z_e,
                                               const float* __restrict__ cb,
                                               const __bf16* __restrict__ ws_hilo,
                                               const float* __restrict__ ws_cnorm,
                                               float* __restrict__ out_zq,
                                               float* __restrict__ out_idx,
                                               float* __restrict__ partial) {
    extern __shared__ __align__(16) char smem[];
    __bf16* s_hi = (__bf16*)smem;                 // 65536 B
    __bf16* s_lo = (__bf16*)(smem + 65536);       // 65536 B
    float*  s_cn = (float*)(smem + 131072);       // 2048 B
    int*    s_ix = (int*)(smem + 133120);         // 512 B
    float*  s_rd = (float*)(smem + 133632);       // 1024 B

    const int tid = threadIdx.x;

    // ---- Stage codebook hi+lo (131072 B contiguous) and cnorm into LDS ----
#if defined(HAVE_ASYNC_LDS) && defined(__HIP_DEVICE_COMPILE__)
    {
        // CDNA5 async-to-LDS path: no VGPR staging, tracked by ASYNCcnt.
        const uint4* src = (const uint4*)ws_hilo;
        uint4* dst = (uint4*)smem;
        #pragma unroll
        for (int i = 0; i < 32; ++i) {
            v4i* g = (v4i*)(uintptr_t)(src + (tid + i * 256));   // global src (const stripped)
            v4i* l = (v4i*)(void*)(dst + (tid + i * 256));       // generic LDS dst
            __builtin_amdgcn_global_load_async_to_lds_b128(g, l, 0, 0);
        }
        for (int i = tid; i < 512; i += 256) s_cn[i] = ws_cnorm[i];
#if __has_builtin(__builtin_amdgcn_s_wait_asynccnt)
        __builtin_amdgcn_s_wait_asynccnt(0);
#else
        asm volatile("s_wait_asynccnt 0x0" ::: "memory");
#endif
    }
#else
    {
        const uint4* src = (const uint4*)ws_hilo;
        uint4* dst = (uint4*)smem;
        #pragma unroll
        for (int i = 0; i < 32; ++i) dst[tid + i * 256] = src[tid + i * 256];
        for (int i = tid; i < 512; i += 256) s_cn[i] = ws_cnorm[i];
    }
#endif
    __syncthreads();

    const int wv   = tid >> 5;
    const int lane = tid & 31;
    const int r    = lane & 15;      // row/code within 16-tile
    const int kh   = lane >> 4;      // lane half selects K sub-range
    const int half8 = kh * 8;

    // ---- Load this wave's 16 z-rows as split-bf16 B fragments (K chunks of 32) ----
    // B layout (32x16 bf16): lane n=l&15, j<8 -> K=(l>>4)*8+j, j>=8 -> K=16+(l>>4)*8+(j-8) within chunk.
    const int rows_base = blockIdx.x * 128 + wv * 16;
    const int bImg = rows_base >> 12;              // /4096: image index (block never crosses b)
    const int nb   = (rows_base & 4095) + r;       // h*W + w
    const float* zb = z_e + (size_t)bImg * CHW_ + nb;

    bf16x16 Bh0, Bh1, Bl0, Bl1;
    #pragma unroll
    for (int j = 0; j < 16; ++j) {
        float x0 = zb[(size_t)( 0 + half8 + j + (j >= 8 ? 8 : 0)) * HW_];   // dims  0..31
        float x1 = zb[(size_t)(32 + half8 + j + (j >= 8 ? 8 : 0)) * HW_];   // dims 32..63
        __bf16 h0 = (__bf16)x0;
        __bf16 h1 = (__bf16)x1;
        Bh0[j] = h0; Bl0[j] = (__bf16)(x0 - (float)h0);
        Bh1[j] = h1; Bl1[j] = (__bf16)(x1 - (float)h1);
    }

    // ---- Scan all 512 codes: 32 tiles x (3 split terms x 2 K-chunks) WMMAs ----
    float best = 3.4e38f;
    int   bidx = 0;

    for (int t = 0; t < 32; ++t) {
        const int code = t * 16 + r;
        // A layout (16x32 bf16): lane row r, elements j<8 -> K=kh*8+j, j>=8 -> K=16+kh*8+(j-8)
        const __bf16* ph = s_hi + code * 64 + half8;
        const __bf16* pl = s_lo + code * 64 + half8;
        bf16x16 Ah0 = ld_frag16(ph);        // dims  0..31
        bf16x16 Ah1 = ld_frag16(ph + 32);   // dims 32..63
        bf16x16 Al0 = ld_frag16(pl);
        bf16x16 Al1 = ld_frag16(pl + 32);

        f32x8 acc = {0.f, 0.f, 0.f, 0.f, 0.f, 0.f, 0.f, 0.f};
        acc = __builtin_amdgcn_wmma_f32_16x16x32_bf16(false, Ah0, false, Bh0, (short)0, acc, false, false);
        acc = __builtin_amdgcn_wmma_f32_16x16x32_bf16(false, Ah1, false, Bh1, (short)0, acc, false, false);
        acc = __builtin_amdgcn_wmma_f32_16x16x32_bf16(false, Ah0, false, Bl0, (short)0, acc, false, false);
        acc = __builtin_amdgcn_wmma_f32_16x16x32_bf16(false, Ah1, false, Bl1, (short)0, acc, false, false);
        acc = __builtin_amdgcn_wmma_f32_16x16x32_bf16(false, Al0, false, Bh0, (short)0, acc, false, false);
        acc = __builtin_amdgcn_wmma_f32_16x16x32_bf16(false, Al1, false, Bh1, (short)0, acc, false, false);

        // dist = ||c||^2 - 2*z.c  (row-constant ||z||^2 dropped: argmin-invariant)
        const float* cn = s_cn + t * 16 + half8;
        #pragma unroll
        for (int v = 0; v < 8; ++v) {
            float dist = cn[v] - 2.f * acc[v];
            int   idx  = t * 16 + half8 + v;    // ascending -> strict '<' keeps first min
            if (dist < best) { best = dist; bidx = idx; }
        }
    }

    // Merge lane halves (same z-row n = l&15, codes m%16 in [0,8) vs [8,16))
    {
        float ob = __shfl_xor(best, 16, 32);
        int   oi = __shfl_xor(bidx, 16, 32);
        if (ob < best || (ob == best && oi < bidx)) { best = ob; bidx = oi; }
    }
    if (lane < 16) s_ix[wv * 16 + r] = bidx;
    __syncthreads();

    // ---- Emit indices, gather z_q (NCHW), accumulate fused loss ----
    const int n_base = blockIdx.x * 128;
    if (tid < 128) out_idx[n_base + tid] = (float)s_ix[tid];

    const int row   = tid & 127;
    const int chalf = tid >> 7;                       // covers 2 channels per iteration
    const size_t obase = (size_t)(n_base >> 12) * CHW_ + (size_t)((n_base & 4095) + row);
    const float* crow = cb + s_ix[row] * 64;          // exact fp32 codebook row (L2-hot)
    float lsum = 0.f;
    #pragma unroll 4
    for (int cc = 0; cc < 32; ++cc) {
        int c = cc * 2 + chalf;
        float cv = crow[c];
        size_t a = obase + (size_t)c * HW_;
        float zv = z_e[a];
        out_zq[a] = cv;
        float d = cv - zv;
        lsum = fmaf(d, d, lsum);
    }
    s_rd[tid] = lsum;
    __syncthreads();
    #pragma unroll
    for (int s = 128; s > 0; s >>= 1) {
        if (tid < s) s_rd[tid] += s_rd[tid + s];
        __syncthreads();
    }
    if (tid == 0) partial[blockIdx.x] = s_rd[0];
}

__global__ __launch_bounds__(256) void vq_finalize(const float* __restrict__ partial,
                                                   float* __restrict__ loss_out) {
    __shared__ float s_rd[256];
    int tid = threadIdx.x;
    float s = 0.f;
    for (int i = tid; i < 1024; i += 256) s += partial[i];
    s_rd[tid] = s;
    __syncthreads();
    #pragma unroll
    for (int k = 128; k > 0; k >>= 1) {
        if (tid < k) s_rd[tid] += s_rd[tid + k];
        __syncthreads();
    }
    // loss = mse + 0.25*mse  (straight-through makes both terms the same MSE)
    if (tid == 0) loss_out[0] = s_rd[0] * (1.25f / (float)NUMEL_);
}

extern "C" void kernel_launch(void* const* d_in, const int* in_sizes, int n_in,
                              void* d_out, int out_size, void* d_ws, size_t ws_size,
                              hipStream_t stream) {
    (void)in_sizes; (void)n_in; (void)out_size; (void)ws_size;
    const float* z_e = (const float*)d_in[0];
    const float* cb  = (const float*)d_in[1];
    float* out = (float*)d_out;

    char*   ws      = (char*)d_ws;
    __bf16* cb_hi   = (__bf16*)ws;
    __bf16* cb_lo   = (__bf16*)(ws + 65536);
    float*  cnorm   = (float*)(ws + 131072);
    float*  partial = (float*)(ws + 133120);

    vq_prep_split<<<128, 256, 0, stream>>>(cb, cb_hi, cb_lo);
    vq_prep_norm<<<2, 256, 0, stream>>>(cb, cnorm);

    // d_out layout: [0, NUMEL) z_q | [NUMEL] loss | [NUMEL+1, +NROWS) indices (as float)
    vq_main<<<1024, 256, 134656, stream>>>(z_e, cb, cb_hi, cnorm,
                                           out, out + (NUMEL_ + 1), partial);
    vq_finalize<<<1, 256, 0, stream>>>(partial, out + NUMEL_);
}
